// gridGPTAC_40157944218140
// MI455X (gfx1250) — matrix operand order repaired and checked by reading, not compiled, not run on hardware
//
#include <hip/hip_runtime.h>

// ---------------------------------------------------------------------------
// Model constants (from the reference)
// ---------------------------------------------------------------------------
#define BB   4
#define LL   1024
#define SS   512
#define DD   512
#define NHH  8
#define HDD  64
#define NLL  4
#define AA   1000
#define MM   (BB * LL)   // 4096 tokens

typedef unsigned short u16;
typedef __attribute__((ext_vector_type(8)))  __bf16 v8bf;
typedef __attribute__((ext_vector_type(16))) __bf16 v16bf;
typedef __attribute__((ext_vector_type(8)))  float  v8f;
typedef __attribute__((ext_vector_type(4)))  int    v4i;

#define DEV static __device__ __forceinline__

#if __has_builtin(__builtin_amdgcn_global_load_async_to_lds_b128) && \
    __has_builtin(__builtin_amdgcn_s_wait_asynccnt)
#define USE_ASYNC_LDS 1
#define AS1 __attribute__((address_space(1)))
#define AS3 __attribute__((address_space(3)))
#else
#define USE_ASYNC_LDS 0
#endif

// f32 -> bf16 (round-to-nearest-even), stored as raw u16
DEV u16 f2bf(float f) {
    union { float f; unsigned u; } c; c.f = f;
    unsigned u = c.u;
    u += 0x7FFFu + ((u >> 16) & 1u);
    return (u16)(u >> 16);
}

DEV v8f v8f_zero() {
    v8f r;
#pragma unroll
    for (int i = 0; i < 8; ++i) r[i] = 0.0f;
    return r;
}

// A-operand fragment (16x32 bf16): lane&15 = row; elems[0..15]=K(16*half+0..15).
// rowp points at row start (bf16 units).
DEV v16bf load_a16(const u16* rowp, int k0, int half) {
    return *(const v16bf*)(rowp + k0 + 16 * half);
}

// B-operand fragment (32x16 bf16) from Bt stored [N,K] row-major:
// lane&15 = col n; elems[0..15] = K(16*half + 0..15) -> one contiguous load.
DEV v16bf load_b16(const u16* rowp, int k0, int half) {
    return *(const v16bf*)(rowp + k0 + 16 * half);
}

#define WMMA_BF16(a, b, c) \
    __builtin_amdgcn_wmma_f32_16x16x32_bf16(false, (a), false, (b), (short)0, (c), false, false)

// ---------------------------------------------------------------------------
// Generic WMMA GEMM:  C[M,N] = A[M,K](bf16,row-major) * Bt[N,K](bf16)^T
// Wave tile 32x64, block tile 128x128 (8 waves), K step 32.
// B tile (128 cols x 32 k = 8KB) is double-buffered through LDS via the
// gfx1250 async-to-LDS path (ASYNCcnt) when the toolchain exposes it.
// Epilogue MODE:
//   0: outB[row*p0 + p1 + col] = bf16(acc + bias[col])      (concat into fused)
//   1: v = acc + res[row*N+col]; outF = v; outB = bf16(v)   (token embed + pos)
//   2: bf16 scatter Q/K -> [B,NH,L,HD]   (p0 = L)
//   3: bf16 scatter V   -> [B,NH,HD,L]   (p0 = L)
//   4: outF[row*N+col] = acc + bias[col] + res[row*N+col]   (pre-LN)
//   5: outB[row*N+col] = bf16(relu(acc + bias[col]))        (MLP hidden)
// ---------------------------------------------------------------------------
template <int MODE>
__global__ __launch_bounds__(256)
void gemm_wmma(const u16* __restrict__ A, int lda,
               const u16* __restrict__ Bt, int ldb,
               int M, int N, int K,
               const float* __restrict__ bias,
               const float* __restrict__ res,
               float* __restrict__ outF,
               u16* __restrict__ outB,
               int p0, int p1) {
    const int lane = threadIdx.x & 31;
    const int wave = threadIdx.x >> 5;
    const int half = lane >> 4;
    const int l15  = lane & 15;
    const int wm = wave >> 1;        // 0..3
    const int wn = wave & 1;         // 0..1
    const int m0 = blockIdx.y * 128 + wm * 32;
    const int nblk = blockIdx.x * 128;
    const int n0 = nblk + wn * 64;

    v8f acc[2][4];
#pragma unroll
    for (int i = 0; i < 2; ++i)
#pragma unroll
        for (int j = 0; j < 4; ++j) acc[i][j] = v8f_zero();

    const u16* arow[2];
#pragma unroll
    for (int i = 0; i < 2; ++i) arow[i] = A + (size_t)(m0 + 16 * i + l15) * lda;

#if USE_ASYNC_LDS
    // ---- async double-buffered B staging ---------------------------------
    __shared__ u16 btile[2][128][32];          // 2 x 8KB
    const int tid = threadIdx.x;
    // Each thread moves two 16B chunks per tile: chunk c -> row n=c>>2, part=c&3.
    const int n_a  = (tid) >> 2,       part_a = (tid) & 3;
    const int n_b  = (tid + 256) >> 2, part_b = (tid + 256) & 3;
    const u16* gA = Bt + (size_t)(nblk + n_a) * ldb + part_a * 8;
    const u16* gB = Bt + (size_t)(nblk + n_b) * ldb + part_b * 8;

#define ISSUE_BTILE(k0_, buf_)                                                   \
    do {                                                                         \
        __builtin_amdgcn_global_load_async_to_lds_b128(                          \
            (AS1 v4i*)(u16*)(gA + (k0_)),                                        \
            (AS3 v4i*)&btile[(buf_)][n_a][part_a * 8], 0, 0);                    \
        __builtin_amdgcn_global_load_async_to_lds_b128(                          \
            (AS1 v4i*)(u16*)(gB + (k0_)),                                        \
            (AS3 v4i*)&btile[(buf_)][n_b][part_b * 8], 0, 0);                    \
    } while (0)

    ISSUE_BTILE(0, 0);
    for (int k0 = 0; k0 < K; k0 += 32) {
        const int cur = (k0 >> 5) & 1;
        if (k0 + 32 < K) {
            ISSUE_BTILE(k0 + 32, cur ^ 1);
            __builtin_amdgcn_s_wait_asynccnt(2);   // current tile landed
        } else {
            __builtin_amdgcn_s_wait_asynccnt(0);
        }
        __syncthreads();                           // tile visible to all waves

        v16bf a[2], b[4];
#pragma unroll
        for (int i = 0; i < 2; ++i) a[i] = load_a16(arow[i], k0, half);
#pragma unroll
        for (int j = 0; j < 4; ++j)
            b[j] = *(const v16bf*)(&btile[cur][wn * 64 + 16 * j + l15][16 * half]);
        if (k0 + 32 < K) __builtin_prefetch(arow[0] + k0 + 32, 0, 3);
#pragma unroll
        for (int i = 0; i < 2; ++i)
#pragma unroll
            for (int j = 0; j < 4; ++j) acc[i][j] = WMMA_BF16(a[i], b[j], acc[i][j]);

        __syncthreads();                           // protect buffer reuse
    }
#undef ISSUE_BTILE
#else
    // ---- direct-load fallback --------------------------------------------
    const u16* brow[4];
#pragma unroll
    for (int j = 0; j < 4; ++j) brow[j] = Bt + (size_t)(n0 + 16 * j + l15) * ldb;

    for (int k0 = 0; k0 < K; k0 += 32) {
        v16bf a[2], b[4];
#pragma unroll
        for (int i = 0; i < 2; ++i) a[i] = load_a16(arow[i], k0, half);
#pragma unroll
        for (int j = 0; j < 4; ++j) b[j] = load_b16(brow[j], k0, half);
        if (k0 + 32 < K) {
            __builtin_prefetch(arow[0] + k0 + 32, 0, 3);
            __builtin_prefetch(brow[0] + k0 + 32, 0, 3);
        }
#pragma unroll
        for (int i = 0; i < 2; ++i)
#pragma unroll
            for (int j = 0; j < 4; ++j) acc[i][j] = WMMA_BF16(a[i], b[j], acc[i][j]);
    }
#endif

#pragma unroll
    for (int i = 0; i < 2; ++i)
#pragma unroll
        for (int j = 0; j < 4; ++j)
#pragma unroll
            for (int r = 0; r < 8; ++r) {
                const int row = m0 + 16 * i + 8 * half + r;
                const int col = n0 + 16 * j + l15;
                float v = acc[i][j][r];
                if (MODE == 0) {
                    if (bias) v += bias[col];
                    outB[(size_t)row * p0 + p1 + col] = f2bf(v);
                } else if (MODE == 1) {
                    v += res[(size_t)row * N + col];
                    outF[(size_t)row * N + col] = v;
                    outB[(size_t)row * N + col] = f2bf(v);
                } else if (MODE == 2) {
                    const int bb = row / p0, t = row % p0;
                    const int h = col >> 6, hd = col & 63;
                    outB[(((size_t)bb * NHH + h) * p0 + t) * HDD + hd] = f2bf(v);
                } else if (MODE == 3) {
                    const int bb = row / p0, t = row % p0;
                    const int h = col >> 6, hd = col & 63;
                    outB[(((size_t)bb * NHH + h) * HDD + hd) * p0 + t] = f2bf(v);
                } else if (MODE == 4) {
                    v += bias[col] + res[(size_t)row * N + col];
                    outF[(size_t)row * N + col] = v;
                } else if (MODE == 5) {
                    v += bias[col];
                    v = v > 0.0f ? v : 0.0f;
                    outB[(size_t)row * N + col] = f2bf(v);
                }
            }
}

// ---------------------------------------------------------------------------
// Causal flash attention, one (b,h) per blockIdx.x, 128 queries per block,
// one 16-query tile per wave.  Q,K: [B,NH,L,HD] bf16.  Vt: [B,NH,HD,L] bf16.
// Out: Ocat [M, NH*HD] bf16 (heads concatenated).
// ---------------------------------------------------------------------------
__global__ __launch_bounds__(256)
void attn_kernel(const u16* __restrict__ Q, const u16* __restrict__ Km,
                 const u16* __restrict__ Vt, u16* __restrict__ Ocat, int L) {
    __shared__ u16 pbuf[8][16][32];   // per-wave P tile (16 q x 32 k) bf16

    const int lane = threadIdx.x & 31;
    const int wave = threadIdx.x >> 5;
    const int half = lane >> 4;
    const int l15  = lane & 15;
    const int bh = blockIdx.x;
    const int b = bh >> 3, h = bh & 7;
    const int q0 = blockIdx.y * 128 + wave * 16;

    const u16* Qb = Q  + ((size_t)b * NHH + h) * L * HDD;
    const u16* Kb = Km + ((size_t)b * NHH + h) * L * HDD;
    const u16* Vb = Vt + ((size_t)b * NHH + h) * HDD * L;

    // Q fragments for K-chunks {0..31},{32..63}
    const u16* qrow = Qb + (size_t)(q0 + l15) * HDD;
    v16bf aq0 = load_a16(qrow, 0, half);
    v16bf aq1 = load_a16(qrow, 32, half);

    float mi[8], li[8];
#pragma unroll
    for (int r = 0; r < 8; ++r) { mi[r] = -1e30f; li[r] = 0.0f; }
    v8f o[4];
#pragma unroll
    for (int j = 0; j < 4; ++j) o[j] = v8f_zero();

    const float scale = 0.125f;  // HD^-0.5

    for (int kt = 0; kt <= q0; kt += 32) {
        // ---- scores: two 16-wide key tiles -------------------------------
        v8f s0 = v8f_zero(), s1 = v8f_zero();
        {
            const u16* k0row = Kb + (size_t)(kt + l15) * HDD;
            const u16* k1row = Kb + (size_t)(kt + 16 + l15) * HDD;
            v16bf b00 = load_b16(k0row, 0, half);
            v16bf b01 = load_b16(k0row, 32, half);
            v16bf b10 = load_b16(k1row, 0, half);
            v16bf b11 = load_b16(k1row, 32, half);
            s0 = WMMA_BF16(aq0, b00, s0);
            s0 = WMMA_BF16(aq1, b01, s0);
            s1 = WMMA_BF16(aq0, b10, s1);
            s1 = WMMA_BF16(aq1, b11, s1);
        }
        // ---- online softmax ---------------------------------------------
#pragma unroll
        for (int r = 0; r < 8; ++r) {
            const int qr = q0 + 8 * half + r;
            float x0 = s0[r] * scale;
            float x1 = s1[r] * scale;
            if (kt + l15 > qr)      x0 = -1e30f;
            if (kt + 16 + l15 > qr) x1 = -1e30f;
            float v = fmaxf(x0, x1);
#pragma unroll
            for (int off = 1; off < 16; off <<= 1) v = fmaxf(v, __shfl_xor(v, off));
            const float mnew = fmaxf(mi[r], v);
            const float corr = __expf(mi[r] - mnew);
            const float e0 = __expf(x0 - mnew);
            const float e1 = __expf(x1 - mnew);
            float rs = e0 + e1;
#pragma unroll
            for (int off = 1; off < 16; off <<= 1) rs += __shfl_xor(rs, off);
            li[r] = li[r] * corr + rs;
            mi[r] = mnew;
#pragma unroll
            for (int j = 0; j < 4; ++j) o[j][r] *= corr;
            pbuf[wave][8 * half + r][l15]      = f2bf(e0);
            pbuf[wave][8 * half + r][16 + l15] = f2bf(e1);
        }
        __threadfence_block();   // order LDS stores vs. cross-lane loads
        // ---- P (C-layout -> A-layout via LDS), then P @ V ----------------
        const u16* prow = &pbuf[wave][l15][0];
        v16bf pa = load_a16(prow, 0, half);
#pragma unroll
        for (int j = 0; j < 4; ++j) {
            const u16* vrow = Vb + (size_t)(j * 16 + l15) * L;
            v16bf bv = load_b16(vrow, kt, half);
            o[j] = WMMA_BF16(pa, bv, o[j]);
        }
    }

    float inv[8];
#pragma unroll
    for (int r = 0; r < 8; ++r) inv[r] = 1.0f / li[r];
#pragma unroll
    for (int j = 0; j < 4; ++j)
#pragma unroll
        for (int r = 0; r < 8; ++r) {
            const int row = b * L + q0 + 8 * half + r;
            const int col = h * HDD + j * 16 + l15;
            Ocat[(size_t)row * (NHH * HDD) + col] = f2bf(o[j][r] * inv[r]);
        }
}

// ---------------------------------------------------------------------------
// LayerNorm over D=512, one wave per row.  outB optional (bf16 mirror).
// ---------------------------------------------------------------------------
__global__ __launch_bounds__(256)
void ln_kernel(const float* __restrict__ in, long long ld,
               const float* __restrict__ g, const float* __restrict__ be,
               float* __restrict__ outF, long long ldo,
               u16* __restrict__ outB, int rows) {
    const int lane = threadIdx.x & 31;
    const int wave = threadIdx.x >> 5;
    const int row = blockIdx.x * 8 + wave;
    if (row >= rows) return;
    const float* p = in + (size_t)row * ld;
    float v[16];
    float s = 0.0f;
#pragma unroll
    for (int i = 0; i < 16; ++i) { v[i] = p[i * 32 + lane]; s += v[i]; }
#pragma unroll
    for (int off = 1; off < 32; off <<= 1) s += __shfl_xor(s, off);
    const float mean = s * (1.0f / 512.0f);
    float q = 0.0f;
#pragma unroll
    for (int i = 0; i < 16; ++i) { const float d = v[i] - mean; q += d * d; }
#pragma unroll
    for (int off = 1; off < 32; off <<= 1) q += __shfl_xor(q, off);
    const float rstd = rsqrtf(q * (1.0f / 512.0f) + 1e-5f);
#pragma unroll
    for (int i = 0; i < 16; ++i) {
        const int c = i * 32 + lane;
        const float ov = (v[i] - mean) * rstd * g[c] + be[c];
        outF[(size_t)row * ldo + c] = ov;
        if (outB) outB[(size_t)row * ldo + c] = f2bf(ov);
    }
}

// ---------------------------------------------------------------------------
// Small helper kernels
// ---------------------------------------------------------------------------
__global__ void cvt_f32_bf16(const float* __restrict__ in, u16* __restrict__ out, int n) {
    const int i = blockIdx.x * blockDim.x + threadIdx.x;
    if (i < n) out[i] = f2bf(in[i]);
}

// W[K,N] f32 -> Wt[N,K] bf16
__global__ void pack_wt(const float* __restrict__ W, u16* __restrict__ Wt, int K, int N) {
    const int i = blockIdx.x * blockDim.x + threadIdx.x;
    if (i < K * N) {
        const int n = i / K, k = i % K;
        Wt[i] = f2bf(W[(size_t)k * N + n]);
    }
}

// W[NH][D][HD] f32 -> Wt[NH*HD][D] bf16   (column n = h*HD+hd)
__global__ void pack_qkv(const float* __restrict__ W, u16* __restrict__ Wt) {
    const int i = blockIdx.x * blockDim.x + threadIdx.x;
    if (i < NHH * HDD * DD) {
        const int d = i % DD;
        const int nh = i / DD;
        const int h = nh >> 6, hd = nh & 63;
        Wt[i] = f2bf(W[((size_t)h * DD + d) * HDD + hd]);
    }
}

// fused[m, 512 + j] = bf16(E_act[action[m], j])
__global__ void embed_act(const int* __restrict__ action, const float* __restrict__ Eact,
                          u16* __restrict__ fused) {
    const int i = blockIdx.x * blockDim.x + threadIdx.x;
    if (i < MM * DD) {
        const int m = i >> 9, j = i & 511;
        fused[(size_t)m * (3 * DD) + DD + j] = f2bf(Eact[(size_t)action[m] * DD + j]);
    }
}

// pe[m,j] = E_idx[slot[m],j] + E_time[ts[m],j] + b_traj[j]
__global__ void posemb_k(const int* __restrict__ slot, const int* __restrict__ ts,
                         const float* __restrict__ Eidx, const float* __restrict__ Etime,
                         const float* __restrict__ btraj, float* __restrict__ pe) {
    const int i = blockIdx.x * blockDim.x + threadIdx.x;
    if (i < MM * DD) {
        const int m = i >> 9, j = i & 511;
        pe[i] = Eidx[(size_t)slot[m] * DD + j] + Etime[(size_t)ts[m] * DD + j] + btraj[j];
    }
}

// logits[b,a] = b_lm[a] + sum_d h[b,d] * W_lm[d,a]
__global__ void head_kernel(const float* __restrict__ h, const float* __restrict__ Wlm,
                            const float* __restrict__ blm, float* __restrict__ out) {
    const int i = blockIdx.x * blockDim.x + threadIdx.x;
    if (i >= BB * AA) return;
    const int b = i / AA, a = i % AA;
    float s = blm[a];
    const float* hp = h + (size_t)b * DD;
#pragma unroll 8
    for (int d = 0; d < DD; ++d) s += hp[d] * Wlm[(size_t)d * AA + a];
    out[i] = s;
}

// ---------------------------------------------------------------------------
// Host-side orchestration
// ---------------------------------------------------------------------------
template <int MODE>
static void launch_gemm(hipStream_t s, const u16* A, int lda, const u16* Bt, int ldb,
                        int M, int N, int K, const float* bias, const float* res,
                        float* outF, u16* outB, int p0, int p1) {
    dim3 grid(N / 128, M / 128), blk(256);
    gemm_wmma<MODE><<<grid, blk, 0, s>>>(A, lda, Bt, ldb, M, N, K, bias, res, outF, outB, p0, p1);
}

extern "C" void kernel_launch(void* const* d_in, const int* in_sizes, int n_in,
                              void* d_out, int out_size, void* d_ws, size_t ws_size,
                              hipStream_t stream) {
    const float* prev_state = (const float*)d_in[0];
    const int*   action     = (const int*)d_in[1];
    const float* next_state = (const float*)d_in[2];
    const int*   slot_idx   = (const int*)d_in[3];
    const int*   timestep   = (const int*)d_in[4];
    const float* W_prev = (const float*)d_in[5];
    const float* b_prev = (const float*)d_in[6];
    const float* E_act  = (const float*)d_in[7];
    const float* W_next = (const float*)d_in[8];
    const float* b_next = (const float*)d_in[9];
    const float* W_traj = (const float*)d_in[10];
    const float* b_traj = (const float*)d_in[11];
    const float* E_idx  = (const float*)d_in[12];
    const float* E_time = (const float*)d_in[13];
    const float* Wq     = (const float*)d_in[14];
    const float* Wk     = (const float*)d_in[15];
    const float* Wv     = (const float*)d_in[16];
    const float* W_proj = (const float*)d_in[17];
    const float* b_proj = (const float*)d_in[18];
    const float* ln1_g  = (const float*)d_in[19];
    const float* ln1_b  = (const float*)d_in[20];
    const float* W1     = (const float*)d_in[21];
    const float* b1     = (const float*)d_in[22];
    const float* W2     = (const float*)d_in[23];
    const float* b2     = (const float*)d_in[24];
    const float* ln2_g  = (const float*)d_in[25];
    const float* ln2_b  = (const float*)d_in[26];
    const float* lnf_g  = (const float*)d_in[27];
    const float* lnf_b  = (const float*)d_in[28];
    const float* W_lm   = (const float*)d_in[29];
    const float* b_lm   = (const float*)d_in[30];

    // ---- workspace carve-out (~87 MB) ------------------------------------
    char* w = (char*)d_ws;
    size_t off = 0;
    auto take = [&](size_t bytes) -> char* {
        char* r = w + off;
        off = (off + bytes + 255) & ~(size_t)255;
        return r;
    };
    u16*   sprevB = (u16*)  take((size_t)MM * SS * 2);
    u16*   snextB = (u16*)  take((size_t)MM * SS * 2);
    u16*   fused  = (u16*)  take((size_t)MM * 3 * DD * 2);
    float* pe     = (float*)take((size_t)MM * DD * 4);
    float* x      = (float*)take((size_t)MM * DD * 4);
    u16*   xb     = (u16*)  take((size_t)MM * DD * 2);
    float* preln  = (float*)take((size_t)MM * DD * 4);
    u16*   qb     = (u16*)  take((size_t)MM * DD * 2);
    u16*   kb     = (u16*)  take((size_t)MM * DD * 2);
    u16*   vb     = (u16*)  take((size_t)MM * DD * 2);
    u16*   ocat   = (u16*)  take((size_t)MM * DD * 2);
    u16*   hb     = (u16*)  take((size_t)MM * 4 * DD * 2);
    float* hlast  = (float*)take((size_t)BB * DD * 4);
    u16*   wprevT = (u16*)  take((size_t)DD * SS * 2);
    u16*   wnextT = (u16*)  take((size_t)DD * SS * 2);
    u16*   wtrajT = (u16*)  take((size_t)DD * 3 * DD * 2);
    u16*   wqT    = (u16*)  take((size_t)DD * DD * 2);
    u16*   wkT    = (u16*)  take((size_t)DD * DD * 2);
    u16*   wvT    = (u16*)  take((size_t)DD * DD * 2);
    u16*   wprojT = (u16*)  take((size_t)DD * DD * 2);
    u16*   w1T    = (u16*)  take((size_t)4 * DD * DD * 2);
    u16*   w2T    = (u16*)  take((size_t)DD * 4 * DD * 2);

    const int TPB = 256;
    auto nb = [](size_t n) { return (unsigned)((n + 255) / 256); };

    // ---- stage 0: converts & static packs --------------------------------
    cvt_f32_bf16<<<nb((size_t)MM * SS), TPB, 0, stream>>>(prev_state, sprevB, MM * SS);
    cvt_f32_bf16<<<nb((size_t)MM * SS), TPB, 0, stream>>>(next_state, snextB, MM * SS);
    pack_wt<<<nb((size_t)SS * DD), TPB, 0, stream>>>(W_prev, wprevT, SS, DD);
    pack_wt<<<nb((size_t)SS * DD), TPB, 0, stream>>>(W_next, wnextT, SS, DD);
    pack_wt<<<nb((size_t)3 * DD * DD), TPB, 0, stream>>>(W_traj, wtrajT, 3 * DD, DD);
    embed_act<<<nb((size_t)MM * DD), TPB, 0, stream>>>(action, E_act, fused);
    posemb_k<<<nb((size_t)MM * DD), TPB, 0, stream>>>(slot_idx, timestep, E_idx, E_time, b_traj, pe);

    // ---- token fusion ----------------------------------------------------
    // e_prev -> fused[:, 0:512], e_next -> fused[:, 1024:1536]
    launch_gemm<0>(stream, sprevB, SS, wprevT, SS, MM, DD, SS, b_prev, nullptr,
                   nullptr, fused, 3 * DD, 0);
    launch_gemm<0>(stream, snextB, SS, wnextT, SS, MM, DD, SS, b_next, nullptr,
                   nullptr, fused, 3 * DD, 2 * DD);
    // tokens = fused @ W_traj + (b_traj + E_idx + E_time)   -> x (f32) + xb (bf16)
    launch_gemm<1>(stream, fused, 3 * DD, wtrajT, 3 * DD, MM, DD, 3 * DD, nullptr, pe,
                   x, xb, 0, 0);

    // ---- transformer layers ----------------------------------------------
    for (int l = 0; l < NLL; ++l) {
        pack_qkv<<<nb((size_t)DD * DD), TPB, 0, stream>>>(Wq + (size_t)l * NHH * DD * HDD, wqT);
        pack_qkv<<<nb((size_t)DD * DD), TPB, 0, stream>>>(Wk + (size_t)l * NHH * DD * HDD, wkT);
        pack_qkv<<<nb((size_t)DD * DD), TPB, 0, stream>>>(Wv + (size_t)l * NHH * DD * HDD, wvT);
        pack_wt<<<nb((size_t)DD * DD), TPB, 0, stream>>>(W_proj + (size_t)l * DD * DD, wprojT, DD, DD);
        pack_wt<<<nb((size_t)DD * 4 * DD), TPB, 0, stream>>>(W1 + (size_t)l * DD * 4 * DD, w1T, DD, 4 * DD);
        pack_wt<<<nb((size_t)DD * 4 * DD), TPB, 0, stream>>>(W2 + (size_t)l * 4 * DD * DD, w2T, 4 * DD, DD);

        launch_gemm<2>(stream, xb, DD, wqT, DD, MM, DD, DD, nullptr, nullptr, nullptr, qb, LL, 0);
        launch_gemm<2>(stream, xb, DD, wkT, DD, MM, DD, DD, nullptr, nullptr, nullptr, kb, LL, 0);
        launch_gemm<3>(stream, xb, DD, wvT, DD, MM, DD, DD, nullptr, nullptr, nullptr, vb, LL, 0);

        attn_kernel<<<dim3(BB * NHH, LL / 128), 256, 0, stream>>>(qb, kb, vb, ocat, LL);

        // proj + residual -> preln ; LN1 -> x, xb
        launch_gemm<4>(stream, ocat, DD, wprojT, DD, MM, DD, DD,
                       b_proj + (size_t)l * DD, x, preln, nullptr, 0, 0);
        ln_kernel<<<MM / 8, 256, 0, stream>>>(preln, DD, ln1_g + (size_t)l * DD,
                                              ln1_b + (size_t)l * DD, x, DD, xb, MM);
        // MLP
        launch_gemm<5>(stream, xb, DD, w1T, DD, MM, 4 * DD, DD,
                       b1 + (size_t)l * 4 * DD, nullptr, nullptr, hb, 0, 0);
        launch_gemm<4>(stream, hb, 4 * DD, w2T, 4 * DD, MM, DD, 4 * DD,
                       b2 + (size_t)l * DD, x, preln, nullptr, 0, 0);
        ln_kernel<<<MM / 8, 256, 0, stream>>>(preln, DD, ln2_g + (size_t)l * DD,
                                              ln2_b + (size_t)l * DD, x, DD, xb, MM);
    }

    // ---- final LN on last tokens + LM head -------------------------------
    ln_kernel<<<1, 256, 0, stream>>>(x + (size_t)(LL - 1) * DD, (long long)LL * DD,
                                     lnf_g, lnf_b, hlast, DD, nullptr, BB);
    head_kernel<<<nb((size_t)BB * AA), TPB, 0, stream>>>(hlast, W_lm, b_lm, (float*)d_out);

    (void)in_sizes; (void)n_in; (void)out_size; (void)ws_size;
}
